// InitLayer_21062519620178
// MI455X (gfx1250) — compile-verified, matrix operand order checked by reading, not run on hardware
//
#include <hip/hip_runtime.h>
#include <hip/hip_bf16.h>

// ---------------------------------------------------------------------------
// Fused equivariant edge-embedding for MI455X (gfx1250, wave32, WMMA).
// Memory-bound problem (~1.15 GB moved, ~50us floor at 23.3 TB/s), so we fuse
// the entire 4-GEMM chain per edge into one pass using v_wmma_f32_16x16x32_bf16,
// keeping all intermediates in VGPRs/LDS. Weights (~128KB packed bf16) stay in L2.
// ---------------------------------------------------------------------------

typedef __attribute__((ext_vector_type(16))) __bf16        v16bf;
typedef __attribute__((ext_vector_type(8)))  float         v8f;
typedef __attribute__((ext_vector_type(8)))  unsigned int  v8u;

#define R_MAX     5.0f
#define NEDGES    524288
#define NNODES    16384
#define K0        160      // 136 padded to 5 K-tiles of 32
#define K0_REAL   136
#define WAVES     4        // 4 wave32 per block, 16 edges per wave

// packed weight fragment offsets in d_ws (units: ushort/bf16 elements)
// layer0: 5x8 frags, layer1: 4x8, layer2: 4x8, env: 4x6 ; 512 elems each
#define OFF_W0    0
#define OFF_W1    20480
#define OFF_W2    36864
#define OFF_WENV  53248
#define WS_ELEMS  65536    // 131072 bytes total

// d_out float offsets (latents, node_features, edge_features, cutoff, active)
#define OUT_LAT   ((size_t)0)
#define OUT_NODE  ((size_t)67108864)          // 524288*128
#define OUT_EDGEF ((size_t)71827456)          // +16384*288
#define OUT_CUT   ((size_t)222822400)         // +524288*288
#define OUT_ACT   ((size_t)223346688)         // +524288

__device__ __forceinline__ unsigned short f2bf(float f) {
  return __builtin_bit_cast(unsigned short, (__bf16)f);
}

// ---------------------------------------------------------------------------
// Kernel 1: pack f32 weights into bf16 WMMA B-fragment layout.
// B fragment (32 K x 16 N): lane l -> n = l%16, half = l/16; element e:
//   vg = e/2, sub = e%2, k = (vg<4?0:16) + (vg%4)*2 + sub + half*8
// Fragment storage: [frag][lane][e], 32B per lane -> 2x b128 loads at use.
// ---------------------------------------------------------------------------
__global__ void pack_weights(const float* __restrict__ W0,
                             const float* __restrict__ W1,
                             const float* __restrict__ W2,
                             const float* __restrict__ We,
                             unsigned short* __restrict__ ws) {
  int idx = blockIdx.x * blockDim.x + threadIdx.x;
  if (idx >= WS_ELEMS) return;
  int frag = idx >> 9;          // 512 elems per fragment
  int rem  = idx & 511;
  int lane = rem >> 4;
  int e    = rem & 15;

  int NT, Kreal, Ncols;
  const float* W;
  if (frag < 40)       {            NT = 8; W = W0; Kreal = 136; Ncols = 128; }
  else if (frag < 72)  { frag -= 40; NT = 8; W = W1; Kreal = 128; Ncols = 128; }
  else if (frag < 104) { frag -= 72; NT = 8; W = W2; Kreal = 128; Ncols = 128; }
  else                 { frag -= 104; NT = 6; W = We; Kreal = 128; Ncols = 96; }

  int kt = frag / NT, nt = frag % NT;
  int half = lane >> 4;
  int n = nt * 16 + (lane & 15);
  int vg = e >> 1, sub = e & 1;
  int k = kt * 32 + ((vg < 4) ? 0 : 16) + (vg & 3) * 2 + sub + half * 8;
  float val = (k < Kreal) ? W[k * Ncols + n] : 0.0f;
  ws[idx] = f2bf(val);
}

// ---------------------------------------------------------------------------
// WMMA helpers
// ---------------------------------------------------------------------------
// A fragment (16 M x 32 K, bf16) read from LDS activation rows:
// lane l -> m = l%16, half = l/16; vg pairs of consecutive k -> ds_load_b32.
__device__ __forceinline__ v16bf load_a_frag(const unsigned short* act, int lane, int kt) {
  int m = lane & 15;
  int half8 = (lane >> 4) << 3;
  const unsigned short* row = act + m * K0;
  v8u u;
#pragma unroll
  for (int vg = 0; vg < 8; vg++) {
    int k = kt * 32 + ((vg < 4) ? 0 : 16) + (vg & 3) * 2 + half8;
    u[vg] = *(const unsigned int*)(row + k);    // 4B aligned (k even)
  }
  return __builtin_bit_cast(v16bf, u);
}

__device__ __forceinline__ v16bf load_b_frag(const unsigned short* ws, int frag, int lane) {
  return __builtin_bit_cast(v16bf, *(const v8u*)(ws + frag * 512 + lane * 16));
}

template <int KT, int NT>
__device__ __forceinline__ void gemm_tile(v8f (&acc)[NT],
                                          const unsigned short* act,
                                          const unsigned short* ws, int lane) {
#pragma unroll
  for (int nt = 0; nt < NT; nt++) acc[nt] = v8f{0.f,0.f,0.f,0.f,0.f,0.f,0.f,0.f};
#pragma unroll
  for (int kt = 0; kt < KT; kt++) {
    v16bf a = load_a_frag(act, lane, kt);
#pragma unroll
    for (int nt = 0; nt < NT; nt++) {
      v16bf b = load_b_frag(ws, kt * NT + nt, lane);
      acc[nt] = __builtin_amdgcn_wmma_f32_16x16x32_bf16(
          false, a, false, b, (short)0, acc[nt], false, false);
    }
  }
}

// silu + bf16, D-layout -> LDS A rows. C/D: vgpr r, lane l => m=r+8*(l/16), n=nt*16+l%16
template <int NT>
__device__ __forceinline__ void store_act_silu(const v8f (&acc)[NT],
                                               unsigned short* act, int lane) {
  int half8 = (lane >> 4) << 3;
  int nl = lane & 15;
#pragma unroll
  for (int nt = 0; nt < NT; nt++)
#pragma unroll
    for (int r = 0; r < 8; r++) {
      float x = acc[nt][r];
      float s = x / (1.0f + __expf(-x));
      act[(r + half8) * K0 + nt * 16 + nl] = f2bf(s);
    }
}

// ---------------------------------------------------------------------------
// Kernel 2: fused per-edge pipeline. 4 waves/block, 16 edges/wave.
// ---------------------------------------------------------------------------
__global__ __launch_bounds__(WAVES * 32) void fused_edge_kernel(
    const int*   __restrict__ edge_index,    // (2,E) row-major; row 0 = center
    const float* __restrict__ edge_sh,       // (E,9)
    const float* __restrict__ edge_length,   // (E)
    const float* __restrict__ edge_one_hot,  // (E,128)
    const float* __restrict__ bessel_w,      // (8)
    const unsigned short* __restrict__ wsW,  // packed bf16 fragments
    float* __restrict__ out_lat,             // (E,128)
    float* __restrict__ out_node,            // (N,288) pre-zeroed
    float* __restrict__ out_edgef,           // (E,288)
    float* __restrict__ out_cut,             // (E)
    float* __restrict__ out_active)          // (E)
{
  __shared__ unsigned short s_act[WAVES][16][K0];  // 20480 B, bf16 activations
  __shared__ float s_w  [WAVES][16][96];           // 24576 B, env weights
  __shared__ float s_sh [WAVES][16][12];           // 3072 B
  __shared__ float s_cut[WAVES][16];
  __shared__ int   s_ctr[WAVES][16];

  const int wave = threadIdx.x >> 5;
  const int lane = threadIdx.x & 31;
  const int e0   = (blockIdx.x * WAVES + wave) * 16;
  unsigned short* act = &s_act[wave][0][0];

  // ---- phase 0: stage inputs -> LDS -------------------------------------
  {
    int row = lane & 15;
    int cb  = (lane >> 4) * 64;
    const float* src = edge_one_hot + (size_t)(e0 + row) * 128 + cb;
    unsigned short* dst = act + row * K0 + cb;
#pragma unroll
    for (int c = 0; c < 64; c += 4) {
      float4 v = *(const float4*)(src + c);
      dst[c + 0] = f2bf(v.x); dst[c + 1] = f2bf(v.y);
      dst[c + 2] = f2bf(v.z); dst[c + 3] = f2bf(v.w);
    }
  }
  if (lane < 16) {
    int e = e0 + lane;
    float r    = edge_length[e];
    float invr = 1.0f / r;
#pragma unroll
    for (int b = 0; b < 8; b++) {
      float arg = bessel_w[b] * (r * (1.0f / R_MAX));
      act[lane * K0 + 128 + b] = f2bf((2.0f / R_MAX) * __sinf(arg) * invr);
    }
#pragma unroll
    for (int k = K0_REAL; k < K0; k++) act[lane * K0 + k] = 0;  // K padding

    float x = r * (1.0f / R_MAX);
    float x2 = x * x, x6 = x2 * x2 * x2;
    float cut = 1.0f - 28.0f * x6 + 48.0f * x6 * x - 21.0f * x6 * x2;
    cut = (x < 1.0f) ? cut : 0.0f;
    float msk = (cut > 0.0f) ? 1.0f : 0.0f;
    s_cut[wave][lane] = cut * msk;
    s_ctr[wave][lane] = edge_index[e];          // edge_center
    out_cut[e]    = cut;
    out_active[e] = (float)e;
#pragma unroll
    for (int j = 0; j < 9; j++) s_sh[wave][lane][j] = edge_sh[(size_t)e * 9 + j];
  }
  __syncthreads();

  // ---- MLP chain on WMMA -------------------------------------------------
  v8f acc[8];
  gemm_tile<5, 8>(acc, act, wsW + OFF_W0, lane);     // [16x160]x[160x128]
  __syncthreads();
  store_act_silu<8>(acc, act, lane);
  __syncthreads();
  gemm_tile<4, 8>(acc, act, wsW + OFF_W1, lane);     // [16x128]x[128x128]
  __syncthreads();
  store_act_silu<8>(acc, act, lane);
  __syncthreads();
  gemm_tile<4, 8>(acc, act, wsW + OFF_W2, lane);     // [16x128]x[128x128]
  __syncthreads();
  {  // latents = cutoff*mask * h ; emit + feed env layer
    int half8 = (lane >> 4) << 3;
    int nl = lane & 15;
#pragma unroll
    for (int nt = 0; nt < 8; nt++)
#pragma unroll
      for (int r = 0; r < 8; r++) {
        int m = r + half8;
        float v = acc[nt][r] * s_cut[wave][m];
        out_lat[(size_t)(e0 + m) * 128 + nt * 16 + nl] = v;
        act[m * K0 + nt * 16 + nl] = f2bf(v);
      }
  }
  __syncthreads();

  v8f accE[6];
  gemm_tile<4, 6>(accE, act, wsW + OFF_WENV, lane);  // [16x128]x[128x96]
  {
    int half8 = (lane >> 4) << 3;
    int nl = lane & 15;
#pragma unroll
    for (int nt = 0; nt < 6; nt++)
#pragma unroll
      for (int r = 0; r < 8; r++)
        s_w[wave][r + half8][nt * 16 + nl] = accE[nt][r];
  }
  __syncthreads();

  // ---- SH outer product expansion + scatter-add --------------------------
  // 288 cols: [0,32)=w(:,0)*sh0 ; [32,128)=w(:,1)*sh[1+j] ; [128,288)=w(:,2)*sh[4+j]
  const float inv_sqrt_neigh = 0.17677669529663687f;  // 1/sqrt(32)
#pragma unroll 1
  for (int m = 0; m < 16; m++) {
    int e = e0 + m;
    float msk = (s_cut[wave][m] > 0.0f) ? 1.0f : 0.0f;
    int   ctr = s_ctr[wave][m];
    float* ef = out_edgef + (size_t)e * 288;
    float* nf = out_node + (size_t)ctr * 288;
#pragma unroll
    for (int it = 0; it < 9; it++) {
      int fc = it * 32 + lane;                      // coalesced column
      int i, c, j, off;
      if (fc < 32)       { i = 0; c = fc;             j = 0;             off = 0; }
      else if (fc < 128) { i = 1; c = (fc - 32) / 3;  j = (fc - 32) % 3; off = 1; }
      else               { i = 2; c = (fc - 128) / 5; j = (fc - 128) % 5; off = 4; }
      float v = s_w[wave][m][i * 32 + c] * s_sh[wave][m][off + j] * msk;
      ef[fc] = v;
      atomicAdd(nf + fc, v * inv_sqrt_neigh);
    }
  }
}

// ---------------------------------------------------------------------------
extern "C" void kernel_launch(void* const* d_in, const int* in_sizes, int n_in,
                              void* d_out, int out_size, void* d_ws, size_t ws_size,
                              hipStream_t stream) {
  const int*   edge_index   = (const int*)d_in[0];
  // d_in[1]=atom_type, d_in[2]=bond_type are unused by the reference math
  const float* edge_sh      = (const float*)d_in[3];
  const float* edge_length  = (const float*)d_in[4];
  const float* edge_one_hot = (const float*)d_in[5];
  const float* bessel_w     = (const float*)d_in[6];
  const float* W0           = (const float*)d_in[7];
  const float* W1           = (const float*)d_in[8];
  const float* W2           = (const float*)d_in[9];
  const float* We           = (const float*)d_in[10];

  float* out = (float*)d_out;
  unsigned short* ws = (unsigned short*)d_ws;

  // 1) weights -> bf16 WMMA fragment layout in scratch (L2-resident, 128KB)
  pack_weights<<<WS_ELEMS / 256, 256, 0, stream>>>(W0, W1, W2, We, ws);

  // 2) zero node-feature accumulator region (atomics target)
  hipMemsetAsync(out + OUT_NODE, 0, (size_t)NNODES * 288 * sizeof(float), stream);

  // 3) fused edge pipeline: 8192 blocks x 4 waves x 16 edges = 524288 edges
  fused_edge_kernel<<<NEDGES / (WAVES * 16), WAVES * 32, 0, stream>>>(
      edge_index, edge_sh, edge_length, edge_one_hot, bessel_w, ws,
      out + OUT_LAT, out + OUT_NODE, out + OUT_EDGEF,
      out + OUT_CUT, out + OUT_ACT);
}
